// _RoIOffsetPooling_64622077935894
// MI455X (gfx1250) — matrix operand order repaired and checked by reading, not compiled, not run on hardware
//
#include <hip/hip_runtime.h>

#define RR    256
#define NFR   2
#define CCH   512
#define HH    64
#define WW    64
#define NBIN  49
#define SCALE 0.0625f
#define GAMMA 0.1f

typedef __attribute__((ext_vector_type(16))) _Float16 v16h;
typedef __attribute__((ext_vector_type(8)))  float    v8f;

// ---------------------------------------------------------------- K0: NCHW -> N(HW)C transpose
__global__ void k_transpose(const float* __restrict__ feat, float* __restrict__ featT) {
    int idx = blockIdx.x * 256 + threadIdx.x;        // NFR*CCH*HH*WW elements
    int c    = idx & (CCH - 1);
    int rest = idx >> 9;
    int pix  = rest & (HH * WW - 1);
    int i    = rest >> 12;
    featT[idx] = feat[((i * CCH + c) * (HH * WW)) + pix];
}

// ---------------------------------------------------------------- K1: channel-mean feature map
__global__ void k_cmean(const float* __restrict__ featT, float* __restrict__ fmean) {
    int lane = threadIdx.x & 31;
    int wv   = threadIdx.x >> 5;
    int pix  = blockIdx.x * 8 + wv;                  // NFR*HH*WW pixels
    const float* p = featT + (size_t)pix * CCH;
    float acc = 0.f;
    for (int c = lane; c < CCH; c += 32) acc += p[c];
    #pragma unroll
    for (int o = 16; o; o >>= 1) acc += __shfl_xor(acc, o, 32);
    if (lane == 0) fmean[pix] = acc * (1.0f / CCH);
}

// ---------------------------------------------------------------- bilinear on 64x64 map
__device__ __forceinline__ float bilin64(const float* __restrict__ f, float y, float x) {
    y = fminf(fmaxf(y, 0.f), 63.f);
    x = fminf(fmaxf(x, 0.f), 63.f);
    float y0 = floorf(y), x0 = floorf(x);
    int y0i = (int)y0, x0i = (int)x0;
    int y1i = min(y0i + 1, 63), x1i = min(x0i + 1, 63);
    float wy = y - y0, wx = x - x0;
    float v00 = f[y0i * 64 + x0i], v01 = f[y0i * 64 + x1i];
    float v10 = f[y1i * 64 + x0i], v11 = f[y1i * 64 + x1i];
    return v00 * (1.f - wy) * (1.f - wx) + v01 * (1.f - wy) * wx
         + v10 * wy * (1.f - wx) + v11 * wy * wx;
}

// ---------------------------------------------------------------- K2: pass-1 pooled means -> WMMA A (f16), ROI params
__global__ void k_pool1(const float* __restrict__ fmean, const float* __restrict__ rois,
                        float* __restrict__ roiP, _Float16* __restrict__ Amat) {
    int b = blockIdx.x;                               // row = r*2 + i
    int r = b >> 1, i = b & 1;
    int t = threadIdx.x;                              // 64 threads
    float x1 = rois[r * 5 + 1] * SCALE, y1 = rois[r * 5 + 2] * SCALE;
    float x2 = rois[r * 5 + 3] * SCALE, y2 = rois[r * 5 + 4] * SCALE;
    float roi_w = fmaxf(x2 - x1, 1.f), roi_h = fmaxf(y2 - y1, 1.f);
    float bw = roi_w / 7.f, bh = roi_h / 7.f;
    if (i == 0 && t == 0) {
        float* p = roiP + r * 6;
        p[0] = y1; p[1] = x1; p[2] = bh; p[3] = bw; p[4] = roi_h; p[5] = roi_w;
    }
    _Float16 val = (_Float16)0.f;
    if (t < NBIN) {
        int ph = t / 7, pw = t % 7;
        const float* fm = fmean + i * (HH * WW);
        float acc = 0.f;
        #pragma unroll
        for (int sy = 0; sy < 2; ++sy)
            #pragma unroll
            for (int sx = 0; sx < 2; ++sx) {
                float ys = y1 + ((float)ph + ((float)sy + 0.5f) * 0.5f) * bh;
                float xs = x1 + ((float)pw + ((float)sx + 0.5f) * 0.5f) * bw;
                acc += bilin64(fm, ys, xs);
            }
        val = (_Float16)(acc * 0.25f);
    }
    Amat[b * 64 + t] = val;                           // K padded 49 -> 64 with zeros
}

// ---------------------------------------------------------------- K3: pack fc_w into WMMA B (f16), [K=64][N=112] padded
__global__ void k_prepB(const float* __restrict__ fc_w, _Float16* __restrict__ Bmat) {
    for (int idx = threadIdx.x; idx < 64 * 112; idx += 256) {
        int k = idx / 112, j = idx % 112;
        float v = (k < 49 && j < 98) ? fc_w[j * 49 + k] : 0.f;   // B[k][j] = fc_w[j][k]
        Bmat[idx] = (_Float16)v;
    }
}

// ---------------------------------------------------------------- K4: off[512][98] = A[512x64] * B[64x112] via v_wmma
__global__ void k_gemm_off(const _Float16* __restrict__ Amat, const _Float16* __restrict__ Bmat,
                           const float* __restrict__ fc_b, float* __restrict__ off) {
    int tile = blockIdx.x;                            // 32 * 7 tiles, one wave each
    int mT = tile / 7, nT = tile % 7;
    int lane = threadIdx.x & 31;
    int half = lane >> 4;
    int mr = mT * 16 + (lane & 15);
    int nc = nT * 16 + (lane & 15);
    v8f acc = {};
    #pragma unroll
    for (int kb = 0; kb < 64; kb += 32) {
        v16h a, bf;
        #pragma unroll
        for (int e = 0; e < 16; ++e) {
            int v  = e >> 1;
            // A k order used here (contiguous): lanes 0-15 K=0..15, lanes 16-31 K=16..31, 2 K per VGPR (matches B)
            int kA = 16 * half + 2 * v + (e & 1);
            a[e] = Amat[mr * 64 + kb + kA];
            // B k order used here (same as A): lanes 0-15 K=0..15, lanes 16-31 K=16..31, 2 K per VGPR
            int kB = 16 * half + 2 * v + (e & 1);
            bf[e] = Bmat[(kb + kB) * 112 + nc];
        }
        acc = __builtin_amdgcn_wmma_f32_16x16x32_f16(false, a, false, bf,
                                                     (short)0, acc, false, false);
    }
    #pragma unroll
    for (int j = 0; j < 8; ++j) {
        int m = mT * 16 + j + 8 * half;               // D layout: VGPR j -> M = j + 8*half
        if (nc < 98) off[m * 98 + nc] = (acc[j] + fc_b[nc]) * GAMMA;
    }
}

// ---------------------------------------------------------------- K5: pass-2 sampling over all channels
__global__ void k_pass2(const float* __restrict__ featT, const float* __restrict__ roiP,
                        const float* __restrict__ off, float* __restrict__ out) {
    int b   = blockIdx.x;                             // ri*49 + bin
    int bin = b % NBIN;
    int ri  = b / NBIN;
    int i = ri & 1, r = ri >> 1;
    const float* p = roiP + r * 6;
    float y1 = p[0], x1 = p[1], bh = p[2], bw = p[3], rh = p[4], rw = p[5];
    float dy = off[ri * 98 + bin * 2 + 0] * rh;
    float dx = off[ri * 98 + bin * 2 + 1] * rw;
    int ph = bin / 7, pw = bin % 7;

    int   b00[4], b01[4], b10[4], b11[4];
    float w00[4], w01[4], w10[4], w11[4];
    #pragma unroll
    for (int s = 0; s < 4; ++s) {
        int sy = s >> 1, sx = s & 1;
        float ys = y1 + ((float)ph + ((float)sy + 0.5f) * 0.5f) * bh + dy;
        float xs = x1 + ((float)pw + ((float)sx + 0.5f) * 0.5f) * bw + dx;
        ys = fminf(fmaxf(ys, 0.f), 63.f);
        xs = fminf(fmaxf(xs, 0.f), 63.f);
        float y0 = floorf(ys), x0 = floorf(xs);
        int y0i = (int)y0, x0i = (int)x0;
        int y1i = min(y0i + 1, 63), x1i = min(x0i + 1, 63);
        float wy = ys - y0, wx = xs - x0;
        int fb = i * (HH * WW);
        b00[s] = (fb + y0i * 64 + x0i) << 9;
        b01[s] = (fb + y0i * 64 + x1i) << 9;
        b10[s] = (fb + y1i * 64 + x0i) << 9;
        b11[s] = (fb + y1i * 64 + x1i) << 9;
        w00[s] = (1.f - wy) * (1.f - wx); w01[s] = (1.f - wy) * wx;
        w10[s] = wy * (1.f - wx);         w11[s] = wy * wx;
    }
    int t = threadIdx.x;                              // 256 threads x 2 channels (float2)
    float accx = 0.f, accy = 0.f;
    #pragma unroll
    for (int s = 0; s < 4; ++s) {
        float2 v00 = ((const float2*)(featT + b00[s]))[t];
        float2 v01 = ((const float2*)(featT + b01[s]))[t];
        float2 v10 = ((const float2*)(featT + b10[s]))[t];
        float2 v11 = ((const float2*)(featT + b11[s]))[t];
        accx += w00[s] * v00.x + w01[s] * v01.x + w10[s] * v10.x + w11[s] * v11.x;
        accy += w00[s] * v00.y + w01[s] * v01.y + w10[s] * v10.y + w11[s] * v11.y;
    }
    int c = t * 2;
    long ob = ((long)(r * (NFR * CCH) + i * CCH + c)) * NBIN + bin;
    out[ob]        = accx * 0.25f;
    out[ob + NBIN] = accy * 0.25f;
}

// ----------------------------------------------------------------
extern "C" void kernel_launch(void* const* d_in, const int* in_sizes, int n_in,
                              void* d_out, int out_size, void* d_ws, size_t ws_size,
                              hipStream_t stream) {
    const float* features = (const float*)d_in[0];
    const float* rois     = (const float*)d_in[1];
    const float* fc_w     = (const float*)d_in[2];
    const float* fc_b     = (const float*)d_in[3];
    float* out = (float*)d_out;

    char* ws = (char*)d_ws;
    float*    featT = (float*)ws;                         // 16,777,216 B
    float*    fmean = (float*)(ws + 16777216);            //     32,768 B
    float*    roiP  = (float*)(ws + 16809984);            //      6,144 B
    _Float16* Amat  = (_Float16*)(ws + 16816128);         //     65,536 B
    _Float16* Bmat  = (_Float16*)(ws + 16881664);         //     14,336 B
    float*    off   = (float*)(ws + 16896000);            //    200,704 B

    k_transpose<<<(NFR * CCH * HH * WW) / 256, 256, 0, stream>>>(features, featT);
    k_cmean   <<<(NFR * HH * WW) / 8, 256, 0, stream>>>(featT, fmean);
    k_pool1   <<<RR * NFR, 64, 0, stream>>>(fmean, rois, roiP, Amat);
    k_prepB   <<<1, 256, 0, stream>>>(fc_w, Bmat);
    k_gemm_off<<<32 * 7, 32, 0, stream>>>(Amat, Bmat, fc_b, off);
    k_pass2   <<<RR * NFR * NBIN, 256, 0, stream>>>(featT, roiP, off, out);
}
